// FC_3204045603697
// MI455X (gfx1250) — matrix operand order repaired and checked by reading, not compiled, not run on hardware
//
#include <hip/hip_runtime.h>
#include <hip/hip_bf16.h>

typedef __attribute__((ext_vector_type(16))) _Float16 v16h;
typedef __attribute__((ext_vector_type(8)))  _Float16 v8h;
typedef __attribute__((ext_vector_type(4)))  _Float16 v4h;
typedef __attribute__((ext_vector_type(8)))  float    v8f;

// ---------------------------------------------------------------------------
// Async copy: global (f16, 16B) -> LDS via the async engine (ASYNCcnt).
// LDS destination address = low 32 bits of the generic pointer (shared
// aperture carries the wave-relative LDS offset in addr[31:0], ISA 10.2).
// ---------------------------------------------------------------------------
__device__ __forceinline__ void async_copy_b128(const _Float16* __restrict__ g,
                                                _Float16* l) {
  const unsigned lds_off = (unsigned)(uintptr_t)l;
  asm volatile("global_load_async_to_lds_b128 %0, %1, off"
               :: "v"(lds_off), "v"(g)
               : "memory");
}
__device__ __forceinline__ void wait_async0() {
  asm volatile("s_wait_asynccnt 0x0" ::: "memory");
}

// ---------------------------------------------------------------------------
// Exact top-k threshold per mask matrix: p-th smallest |m| via 31-step bitwise
// binary search on the (monotone for non-negative floats) uint32 encoding.
// kept element  <=>  bits(|m|) > bits(threshold).
// ---------------------------------------------------------------------------
__global__ __launch_bounds__(256) void topk_threshold(
    const float* __restrict__ m1, int n1,
    const float* __restrict__ m2, int n2,
    const float* __restrict__ m3, int n3,
    float* __restrict__ tout) {
  const float* m;
  int n;
  if (blockIdx.x == 0)      { m = m1; n = n1; }
  else if (blockIdx.x == 1) { m = m2; n = n2; }
  else                      { m = m3; n = n3; }
  const long long p = (long long)n * 7 / 10;  // int(0.7*n) (exact for our sizes)

  __shared__ int scnt[256];
  unsigned ans = 0u;
  for (int bit = 30; bit >= 0; --bit) {
    const unsigned cand = ans | (1u << bit);
    int cnt = 0;
    for (int i = threadIdx.x; i < n; i += 256)
      cnt += (__float_as_uint(fabsf(m[i])) < cand) ? 1 : 0;
    scnt[threadIdx.x] = cnt;
    __syncthreads();
    for (int s = 128; s > 0; s >>= 1) {
      if ((int)threadIdx.x < s) scnt[threadIdx.x] += scnt[threadIdx.x + s];
      __syncthreads();
    }
    const int total = scnt[0];
    __syncthreads();
    if ((long long)total < p) ans = cand;  // keep largest v with count_lt(v) < p
  }
  if (threadIdx.x == 0) tout[blockIdx.x] = __uint_as_float(ans);
}

// ---------------------------------------------------------------------------
// Pack masked weight into zero-padded f16 matrix [Opad x Cpad].
// ---------------------------------------------------------------------------
__global__ __launch_bounds__(256) void pack_masked(
    const float* __restrict__ w, const float* __restrict__ m,
    const float* __restrict__ tptr, _Float16* __restrict__ out,
    int O, int C, int Opad, int Cpad) {
  const int idx = blockIdx.x * 256 + threadIdx.x;
  if (idx >= Opad * Cpad) return;
  const int o = idx / Cpad;
  const int k = idx - o * Cpad;
  float v = 0.f;
  const unsigned tu = __float_as_uint(*tptr);
  if (o < O && k < C) {
    const unsigned u = __float_as_uint(fabsf(m[o * C + k]));
    if (u > tu) v = w[o * C + k];
  }
  out[idx] = (_Float16)v;
}

// ---------------------------------------------------------------------------
// Fragment loaders (ISA 7.12.2 layouts, wave32).
// A 16x32 f16: lane L row M=L&15; lo 8 halves K=kA..kA+7, hi 8 K=kA+16..kA+23,
// kA = (L>=16)*8.
// ---------------------------------------------------------------------------
__device__ __forceinline__ v16h load_fragA(const _Float16* base, int r0,
                                           int stride, int lane) {
  const _Float16* row =
      base + (size_t)(r0 + (lane & 15)) * stride + ((lane & 16) ? 8 : 0);
  v8h lo = *(const v8h*)(row);       // K = kA .. kA+7
  v8h hi = *(const v8h*)(row + 16);  // K = kA+16 .. kA+23
  return __builtin_shufflevector(lo, hi, 0, 1, 2, 3, 4, 5, 6, 7, 8, 9, 10, 11,
                                 12, 13, 14, 15);
}

// B 32x16 f16: lane L holds column N=L&15; lo 8 K=kB..kB+7, hi 8 K=kB+16..kB+23,
// kB = (L>=16)*8.
__device__ __forceinline__ v16h load_fragB(const _Float16* base, int c0,
                                           int stride, int lane) {
  const _Float16* p =
      base + (size_t)(c0 + (lane & 15)) * stride + ((lane >> 4) << 3);
  v8h lo = *(const v8h*)(p);
  v8h hi = *(const v8h*)(p + 16);
  return __builtin_shufflevector(lo, hi, 0, 1, 2, 3, 4, 5, 6, 7, 8, 9, 10, 11,
                                 12, 13, 14, 15);
}

// ---------------------------------------------------------------------------
// Fused 3-layer MLP. Each block: 64 rows of x. 8 waves (wave32).
// LDS: h1 region 40KB persistent; 24KB region reused across layers = 64KB.
// Weight chunks staged via async global->LDS copies (ASYNCcnt).
// ---------------------------------------------------------------------------
__global__ __launch_bounds__(256) void mlp_fused(
    const float* __restrict__ x, const _Float16* __restrict__ w1h,
    const _Float16* __restrict__ w2h, const _Float16* __restrict__ w3h,
    float* __restrict__ out) {
  __shared__ _Float16 sh_h1[64 * 320];  // 40960 B
  __shared__ _Float16 sh_B[12288];      // 24576 B, multi-purpose
  // layer1: [0..2047]=X chunk 64x32, [2048..12287]=W1 chunk 320x32
  // layer2: [0..4095]=W2 chunk 128x32, [4096..12287]=h2 64x128
  // layer3: [0..2047]=W3 16x128,       [4096..12287]=h2

  const int tid = threadIdx.x;
  const int lane = tid & 31;
  const int wave = tid >> 5;
  const int row0 = blockIdx.x * 64;
  const int waveM = wave >> 2;  // 0..1 -> 32 rows
  const int waveN = wave & 3;   // 0..3
  const int sub = (lane >> 4) << 3;

  // ---------------- layer 1: h1 = relu(x @ W1^T), [64 x 320], K=800 --------
  v8f acc1[2][5] = {};
  for (int kc = 0; kc < 25; ++kc) {
    // stage W1 chunk (320 rows x 32 k, f16) : async global -> LDS
#pragma unroll
    for (int i = 0; i < 5; ++i) {
      const int c = tid + 256 * i;
      const int r = c >> 2;
      const int ko = (c & 3) << 3;
      async_copy_b128(w1h + (size_t)r * 800 + kc * 32 + ko,
                      sh_B + 2048 + r * 32 + ko);
    }
    // stage X chunk (f32 -> f16 convert) : 64 rows x 32 k
#pragma unroll
    for (int i = 0; i < 2; ++i) {
      const int q = tid + 256 * i;
      const int r = q >> 3;
      const int kq = (q & 7) << 2;
      const int gk = kc * 32 + kq;
      v4h h = {_Float16(0), _Float16(0), _Float16(0), _Float16(0)};
      if (gk < 784) {
        const float4 v = *(const float4*)(x + (size_t)(row0 + r) * 784 + gk);
        h[0] = (_Float16)v.x; h[1] = (_Float16)v.y;
        h[2] = (_Float16)v.z; h[3] = (_Float16)v.w;
      }
      *(v4h*)(sh_B + r * 32 + kq) = h;
    }
    wait_async0();
    __syncthreads();

    v16h a[2];
#pragma unroll
    for (int mt = 0; mt < 2; ++mt)
      a[mt] = load_fragA(sh_B, waveM * 32 + mt * 16, 32, lane);
#pragma unroll
    for (int nt = 0; nt < 5; ++nt) {
      const v16h b = load_fragB(sh_B + 2048, waveN * 80 + nt * 16, 32, lane);
#pragma unroll
      for (int mt = 0; mt < 2; ++mt)
        acc1[mt][nt] = __builtin_amdgcn_wmma_f32_16x16x32_f16(
            false, a[mt], false, b, (short)0, acc1[mt][nt], false, false);
    }
    __syncthreads();
  }
  // relu + f16 store into sh_h1 (row-major, stride 320)
#pragma unroll
  for (int mt = 0; mt < 2; ++mt)
#pragma unroll
    for (int nt = 0; nt < 5; ++nt) {
      const int rr = waveM * 32 + mt * 16 + sub;
      const int col = waveN * 80 + nt * 16 + (lane & 15);
#pragma unroll
      for (int r = 0; r < 8; ++r)
        sh_h1[(rr + r) * 320 + col] = (_Float16)fmaxf(acc1[mt][nt][r], 0.f);
    }
  __syncthreads();

  // ---------------- layer 2: h2 = relu(h1 @ W2^T), [64 x 128], K=320 -------
  v8f acc2[2][2] = {};
  for (int kc = 0; kc < 10; ++kc) {
#pragma unroll
    for (int i = 0; i < 2; ++i) {
      const int c = tid + 256 * i;
      const int r = c >> 2;
      const int ko = (c & 3) << 3;
      async_copy_b128(w2h + (size_t)r * 320 + kc * 32 + ko,
                      sh_B + r * 32 + ko);
    }
    wait_async0();
    __syncthreads();
    v16h a[2];
#pragma unroll
    for (int mt = 0; mt < 2; ++mt)
      a[mt] = load_fragA(sh_h1 + kc * 32, waveM * 32 + mt * 16, 320, lane);
#pragma unroll
    for (int nt = 0; nt < 2; ++nt) {
      const v16h b = load_fragB(sh_B, waveN * 32 + nt * 16, 32, lane);
#pragma unroll
      for (int mt = 0; mt < 2; ++mt)
        acc2[mt][nt] = __builtin_amdgcn_wmma_f32_16x16x32_f16(
            false, a[mt], false, b, (short)0, acc2[mt][nt], false, false);
    }
    __syncthreads();
  }
#pragma unroll
  for (int mt = 0; mt < 2; ++mt)
#pragma unroll
    for (int nt = 0; nt < 2; ++nt) {
      const int rr = waveM * 32 + mt * 16 + sub;
      const int col = waveN * 32 + nt * 16 + (lane & 15);
#pragma unroll
      for (int r = 0; r < 8; ++r)
        sh_B[4096 + (rr + r) * 128 + col] =
            (_Float16)fmaxf(acc2[mt][nt][r], 0.f);
    }
  __syncthreads();

  // stage W3 (16 x 128 f16, whole) : async global -> LDS
  {
    const int r = tid >> 4;
    const int ko = (tid & 15) << 3;
    async_copy_b128(w3h + r * 128 + ko, sh_B + r * 128 + ko);
  }
  wait_async0();
  __syncthreads();

  // ---------------- layer 3: out = h2 @ W3^T, [64 x 16], K=128 -------------
  if (wave < 4) {
    v8f acc3 = {};
#pragma unroll
    for (int kc = 0; kc < 4; ++kc) {
      const v16h a = load_fragA(sh_B + 4096 + kc * 32, wave * 16, 128, lane);
      const v16h b = load_fragB(sh_B + kc * 32, 0, 128, lane);
      acc3 = __builtin_amdgcn_wmma_f32_16x16x32_f16(false, a, false, b,
                                                    (short)0, acc3, false,
                                                    false);
    }
    const int col = lane & 15;
    const int rr = row0 + wave * 16 + sub;
    if (col < 10) {
#pragma unroll
      for (int r = 0; r < 8; ++r)
        out[(size_t)(rr + r) * 10 + col] = acc3[r];
    }
  }
}

// ---------------------------------------------------------------------------
extern "C" void kernel_launch(void* const* d_in, const int* in_sizes, int n_in,
                              void* d_out, int out_size, void* d_ws,
                              size_t ws_size, hipStream_t stream) {
  const float* x  = (const float*)d_in[0];
  const float* w1 = (const float*)d_in[1];
  const float* m1 = (const float*)d_in[2];
  const float* w2 = (const float*)d_in[3];
  const float* m2 = (const float*)d_in[4];
  const float* w3 = (const float*)d_in[5];
  const float* m3 = (const float*)d_in[6];
  float* out = (float*)d_out;

  char* ws = (char*)d_ws;
  float* thr = (float*)ws;                                    // 3 floats
  _Float16* w1h = (_Float16*)(ws + 256);                      // 320x800 f16
  _Float16* w2h = (_Float16*)(ws + 256 + 512000);             // 128x320 f16
  _Float16* w3h = (_Float16*)(ws + 256 + 512000 + 81920);     // 16x128  f16

  const int n1 = in_sizes[2];  // 235200
  const int n2 = in_sizes[4];  // 30000
  const int n3 = in_sizes[6];  // 1000
  const int B = in_sizes[0] / 784;  // 65536

  topk_threshold<<<3, 256, 0, stream>>>(m1, n1, m2, n2, m3, n3, thr);
  pack_masked<<<(320 * 800 + 255) / 256, 256, 0, stream>>>(
      w1, m1, thr + 0, w1h, 300, 784, 320, 800);
  pack_masked<<<(128 * 320 + 255) / 256, 256, 0, stream>>>(
      w2, m2, thr + 1, w2h, 100, 300, 128, 320);
  pack_masked<<<(16 * 128 + 255) / 256, 256, 0, stream>>>(
      w3, m3, thr + 2, w3h, 10, 100, 16, 128);
  mlp_fused<<<B / 64, 256, 0, stream>>>(x, w1h, w2h, w3h, out);
}
